// TorsionAttention_52999896433133
// MI455X (gfx1250) — matrix operand order repaired and checked by reading, not compiled, not run on hardware
//
#include <hip/hip_runtime.h>
#include <stdint.h>

// ---------------------------------------------------------------------------
// Torsion attention for MI455X (gfx1250, wave32).
//  * All matmuls: V_WMMA_F32_16X16X32_BF16, f32 accumulate.
//  * LDS tile staging: GLOBAL_LOAD_ASYNC_TO_LDS_B128 (ASYNCcnt), double
//    buffered; weights pre-transposed so every stage is contiguous 16B.
//  * Softmax reductions: DPP16 row_xmask butterflies (no LDS bpermute).
// ---------------------------------------------------------------------------

#define TA_B   2
#define TA_S   2048
#define TA_H   1024
#define TA_NH  16
#define TA_HD  64
#define TA_R   64
#define TA_M   (TA_B * TA_S)   // 4096 rows

typedef __attribute__((ext_vector_type(16))) __bf16 v16bf;
typedef __attribute__((ext_vector_type(8)))  float  v8f;

union BF16Frag {
    v16bf v;
    uint4 q[2];
    unsigned short u[16];
};

__device__ __forceinline__ unsigned short f32_to_bf16(float f) {
    union { float f; unsigned u; } x; x.f = f;
    unsigned r = x.u + 0x7FFFu + ((x.u >> 16) & 1u);   // round-to-nearest-even
    return (unsigned short)(r >> 16);
}

__device__ __forceinline__ v16bf load_frag2(const unsigned short* p0,
                                            const unsigned short* p1) {
    BF16Frag t;
    t.q[0] = *reinterpret_cast<const uint4*>(p0);
    t.q[1] = *reinterpret_cast<const uint4*>(p1);
    return t.v;
}

__device__ __forceinline__ v8f wmma_bf16(v16bf a, v16bf b, v8f c) {
    return __builtin_amdgcn_wmma_f32_16x16x32_bf16(
        false, a, false, b, (short)0, c, false, false);
}

// Generic pointer to a __shared__ object: low 32 bits are the LDS byte address
// (ISA 10.2).
__device__ __forceinline__ unsigned lds_addr(const void* p) {
    return (unsigned)(uintptr_t)p;
}

// CDNA5 async copy global -> LDS, 16B per lane, tracked by ASYNCcnt.
__device__ __forceinline__ void async_b128(unsigned lds, const void* gptr) {
    asm volatile("global_load_async_to_lds_b128 %0, %1, off"
                 :: "v"(lds), "v"((unsigned long long)(uintptr_t)gptr)
                 : "memory");
}
__device__ __forceinline__ void wait_async0() {
    asm volatile("s_wait_asynccnt 0x0" ::: "memory");
}
__device__ __forceinline__ void wait_ds0() {
    asm volatile("s_wait_dscnt 0x0" ::: "memory");
}

// DPP16 row_xmask butterfly within a 16-lane row (dpp_ctrl 0x160+mask).
template<int MASK>
__device__ __forceinline__ float dpp_xor16(float x) {
    int r = __builtin_amdgcn_update_dpp(
        0, __float_as_int(x), 0x160 + MASK, 0xf, 0xf, true);
    return __int_as_float(r);
}
__device__ __forceinline__ float rowmax16(float v) {
    v = fmaxf(v, dpp_xor16<1>(v));
    v = fmaxf(v, dpp_xor16<2>(v));
    v = fmaxf(v, dpp_xor16<4>(v));
    v = fmaxf(v, dpp_xor16<8>(v));
    return v;
}
__device__ __forceinline__ float rowsum16(float v) {
    v += dpp_xor16<1>(v);
    v += dpp_xor16<2>(v);
    v += dpp_xor16<4>(v);
    v += dpp_xor16<8>(v);
    return v;
}

// ---------------------------------------------------------------------------
// Elementwise fp32 -> bf16 cast (natural layout)
// ---------------------------------------------------------------------------
__global__ void ta_cast_f32_bf16(const float* __restrict__ src,
                                 unsigned short* __restrict__ dst, int n) {
    int i = blockIdx.x * blockDim.x + threadIdx.x;
    if (i < n) dst[i] = f32_to_bf16(src[i]);
}

// Cast + transpose: src [K][N] f32 -> dst [N][K] bf16 (weights, done once).
__global__ void ta_cast_transpose(const float* __restrict__ src,
                                  unsigned short* __restrict__ dst,
                                  int K, int N) {
    int idx = blockIdx.x * blockDim.x + threadIdx.x;
    if (idx >= K * N) return;
    int k = idx / N, n = idx - k * N;          // coalesced read
    dst[(size_t)n * K + k] = f32_to_bf16(src[idx]);
}

// ---------------------------------------------------------------------------
// Tiled bf16 GEMM: C[M,N] = A[M,K] @ B[K,N] (+bias), B given TRANSPOSED as
// Bt[N,K]. Block tile 128x64, BK=32, double-buffered async LDS staging.
// 8 waves in 4(M)x2(N) grid, each wave = 2x2 WMMA 16x16 tiles.
// Requires M%128==0, N%64==0, K%32==0 (true for all uses here).
// ---------------------------------------------------------------------------
#define GBM 128
#define GBN 64
#define GBK 32

template<bool HAS_BIAS, bool WRITE_F, bool WRITE_B, bool WRITE_BT>
__global__ __launch_bounds__(256) void ta_gemm_bf16(
    const unsigned short* __restrict__ A,    // [M][K]
    const unsigned short* __restrict__ Bt,   // [N][K]
    const float* __restrict__ bias,          // [N]
    float* __restrict__ Cf,                  // [M][N]
    unsigned short* __restrict__ Cb,         // [M][N]
    unsigned short* __restrict__ Cbt,        // [N][M]  (pre-transposed out)
    int M, int N, int K)
{
    __shared__ __align__(16) unsigned short As[2][GBM * GBK];  // [m][k]
    __shared__ __align__(16) unsigned short Bs[2][GBN * GBK];  // [n][k]

    const int tid    = threadIdx.x;
    const int wave   = tid >> 5;
    const int lane   = tid & 31;
    const int laneM  = lane & 15;
    const int laneHi = lane >> 4;
    const int m_blk  = blockIdx.x * GBM;
    const int n_blk  = blockIdx.y * GBN;
    const int wm     = (wave & 3) * 32;
    const int wn     = (wave >> 2) * 32;

    // per-thread staging coordinates (16B granules)
    const int ar0 = tid >> 2,          ac0 = (tid & 3) * 8;   // A rows 0..63
    const int ar1 = (tid + 256) >> 2;                          // A rows 64..127
    const int bn  = tid >> 2,          bc  = (tid & 3) * 8;   // Bt rows 0..63

    auto stage = [&](int buf, int k0) {
        async_b128(lds_addr(&As[buf][ar0 * GBK + ac0]),
                   &A[(size_t)(m_blk + ar0) * K + k0 + ac0]);
        async_b128(lds_addr(&As[buf][ar1 * GBK + ac0]),
                   &A[(size_t)(m_blk + ar1) * K + k0 + ac0]);
        async_b128(lds_addr(&Bs[buf][bn * GBK + bc]),
                   &Bt[(size_t)(n_blk + bn) * K + k0 + bc]);
    };

    v8f acc[2][2];
    #pragma unroll
    for (int i = 0; i < 2; ++i)
        #pragma unroll
        for (int j = 0; j < 2; ++j) acc[i][j] = (v8f){};

    stage(0, 0);
    int buf = 0;
    for (int k0 = 0; k0 < K; k0 += GBK) {
        wait_async0();
        __syncthreads();                       // tile k0 visible to all waves
        if (k0 + GBK < K) stage(buf ^ 1, k0 + GBK);

        v16bf afr[2], bfr[2];
        #pragma unroll
        for (int i = 0; i < 2; ++i) {
            const unsigned short* base = &As[buf][(wm + i * 16 + laneM) * GBK];
            afr[i] = load_frag2(base + laneHi * 8, base + 16 + laneHi * 8);
        }
        #pragma unroll
        for (int j = 0; j < 2; ++j) {
            const unsigned short* base =
                &Bs[buf][(wn + j * 16 + laneM) * GBK];
            bfr[j] = load_frag2(base + laneHi * 8, base + 16 + laneHi * 8);
        }
        #pragma unroll
        for (int i = 0; i < 2; ++i)
            #pragma unroll
            for (int j = 0; j < 2; ++j)
                acc[i][j] = wmma_bf16(afr[i], bfr[j], acc[i][j]);

        __syncthreads();                       // done reading before overwrite
        buf ^= 1;
    }

    // epilogue: C layout VGPR r -> M=laneHi*8+r, N=laneM (compile-time paths)
    #pragma unroll
    for (int i = 0; i < 2; ++i) {
        #pragma unroll
        for (int j = 0; j < 2; ++j) {
            int n = n_blk + wn + j * 16 + laneM;
            float bv = HAS_BIAS ? bias[n] : 0.0f;
            #pragma unroll
            for (int r = 0; r < 8; ++r) {
                int m = m_blk + wm + i * 16 + laneHi * 8 + r;
                float v = acc[i][j][r] + bv;
                if constexpr (WRITE_F)  Cf[(size_t)m * N + n] = v;
                if constexpr (WRITE_B)  Cb[(size_t)m * N + n] = f32_to_bf16(v);
                if constexpr (WRITE_BT) Cbt[(size_t)n * M + m] = f32_to_bf16(v);
            }
        }
    }
}

// ---------------------------------------------------------------------------
// Fused torsion correction: given Lo[M][128] (cols 0-63: low order0,
// cols 64-127: low order1, bf16) and per-order tb^T [H][R], compute
//   l_n = low_n @ tb_n   (K = R = 64, via WMMA)
//   out = bf16( x + sigmoid(c) * (sin(2*pi*l0)*l0 + sin(4*pi*l1)*l1/2) )
// Never materializes lin in HBM. Same 128x64 block tiling as the GEMM.
// ---------------------------------------------------------------------------
__global__ __launch_bounds__(256) void ta_torsion_fused(
    const unsigned short* __restrict__ Lo,   // [M][2R]
    const unsigned short* __restrict__ Bt0,  // [H][R]
    const unsigned short* __restrict__ Bt1,  // [H][R]
    const float* __restrict__ x,             // [M][H] f32
    const float* __restrict__ coupling,      // scalar
    unsigned short* __restrict__ outb)       // [M][H] bf16
{
    __shared__ __align__(16) unsigned short As0[2][GBM * GBK];
    __shared__ __align__(16) unsigned short As1[2][GBM * GBK];
    __shared__ __align__(16) unsigned short Bs0[2][GBN * GBK];
    __shared__ __align__(16) unsigned short Bs1[2][GBN * GBK];

    const int tid    = threadIdx.x;
    const int wave   = tid >> 5;
    const int lane   = tid & 31;
    const int laneM  = lane & 15;
    const int laneHi = lane >> 4;
    const int m_blk  = blockIdx.x * GBM;
    const int n_blk  = blockIdx.y * GBN;
    const int wm     = (wave & 3) * 32;
    const int wn     = (wave >> 2) * 32;

    const int ar0 = tid >> 2,         ac0 = (tid & 3) * 8;
    const int ar1 = (tid + 256) >> 2;
    const int bn  = tid >> 2,         bc  = (tid & 3) * 8;

    auto stage = [&](int buf, int k0) {
        // Lo rows are 128 wide: order0 at +0, order1 at +64
        async_b128(lds_addr(&As0[buf][ar0 * GBK + ac0]),
                   &Lo[(size_t)(m_blk + ar0) * (2 * TA_R) + k0 + ac0]);
        async_b128(lds_addr(&As0[buf][ar1 * GBK + ac0]),
                   &Lo[(size_t)(m_blk + ar1) * (2 * TA_R) + k0 + ac0]);
        async_b128(lds_addr(&As1[buf][ar0 * GBK + ac0]),
                   &Lo[(size_t)(m_blk + ar0) * (2 * TA_R) + TA_R + k0 + ac0]);
        async_b128(lds_addr(&As1[buf][ar1 * GBK + ac0]),
                   &Lo[(size_t)(m_blk + ar1) * (2 * TA_R) + TA_R + k0 + ac0]);
        async_b128(lds_addr(&Bs0[buf][bn * GBK + bc]),
                   &Bt0[(size_t)(n_blk + bn) * TA_R + k0 + bc]);
        async_b128(lds_addr(&Bs1[buf][bn * GBK + bc]),
                   &Bt1[(size_t)(n_blk + bn) * TA_R + k0 + bc]);
    };

    v8f acc0[2][2], acc1[2][2];
    #pragma unroll
    for (int i = 0; i < 2; ++i)
        #pragma unroll
        for (int j = 0; j < 2; ++j) { acc0[i][j] = (v8f){}; acc1[i][j] = (v8f){}; }

    stage(0, 0);
    int buf = 0;
    #pragma unroll
    for (int k0 = 0; k0 < TA_R; k0 += GBK) {   // 2 steps
        wait_async0();
        __syncthreads();
        if (k0 + GBK < TA_R) stage(buf ^ 1, k0 + GBK);

        #pragma unroll
        for (int ord = 0; ord < 2; ++ord) {
            const unsigned short* Asb = ord ? As1[buf] : As0[buf];
            const unsigned short* Bsb = ord ? Bs1[buf] : Bs0[buf];
            v16bf afr[2], bfr[2];
            #pragma unroll
            for (int i = 0; i < 2; ++i) {
                const unsigned short* base = &Asb[(wm + i * 16 + laneM) * GBK];
                afr[i] = load_frag2(base + laneHi * 8, base + 16 + laneHi * 8);
            }
            #pragma unroll
            for (int j = 0; j < 2; ++j) {
                const unsigned short* base =
                    &Bsb[(wn + j * 16 + laneM) * GBK];
                bfr[j] = load_frag2(base + laneHi * 8, base + 16 + laneHi * 8);
            }
            #pragma unroll
            for (int i = 0; i < 2; ++i)
                #pragma unroll
                for (int j = 0; j < 2; ++j) {
                    v8f* acc = ord ? &acc1[i][j] : &acc0[i][j];
                    *acc = wmma_bf16(afr[i], bfr[j], *acc);
                }
        }
        __syncthreads();
        buf ^= 1;
    }

    // torsion epilogue
    const float TWO_PI = 6.283185307179586f;
    float c   = coupling[0];
    float sig = 1.0f / (1.0f + __expf(-c));
    #pragma unroll
    for (int i = 0; i < 2; ++i) {
        #pragma unroll
        for (int j = 0; j < 2; ++j) {
            int n = n_blk + wn + j * 16 + laneM;
            #pragma unroll
            for (int r = 0; r < 8; ++r) {
                int m = m_blk + wm + i * 16 + laneHi * 8 + r;
                float l0 = acc0[i][j][r];
                float l1 = acc1[i][j][r];
                float corr = __sinf(TWO_PI * l0) * l0
                           + 0.5f * __sinf(2.0f * TWO_PI * l1) * l1;
                float xv = x[(size_t)m * TA_H + n];
                outb[(size_t)m * TA_H + n] = f32_to_bf16(xv + sig * corr);
            }
        }
    }
}

// ---------------------------------------------------------------------------
// Flash attention. Grid: x = 128-query tile, y = b*NH + h. 8 waves x 16 rows.
// K chunk [32key][64d] and V chunk [64d][32key] (from pre-transposed Vt[H][M])
// staged with double-buffered async b128 loads. Per-wave P tile round-trips
// through LDS (C layout -> WMMA A layout). Softmax reductions via DPP16.
// ---------------------------------------------------------------------------
#define AKC 32

__global__ __launch_bounds__(256) void ta_flash_attn(
    const unsigned short* __restrict__ Qb,   // [M][H] bf16
    const unsigned short* __restrict__ Kb,   // [M][H] bf16
    const unsigned short* __restrict__ Vt,   // [H][M] bf16 (transposed V)
    unsigned short* __restrict__ Ob)         // [M][H] bf16
{
    __shared__ __align__(16) unsigned short Ks[2][AKC * TA_HD];  // [key][d]
    __shared__ __align__(16) unsigned short Vs[2][TA_HD * AKC];  // [d][key]
    __shared__ __align__(16) unsigned short Ps[8][16 * AKC];     // per-wave P

    const int tid    = threadIdx.x;
    const int wave   = tid >> 5;
    const int lane   = tid & 31;
    const int laneM  = lane & 15;
    const int laneHi = lane >> 4;
    const int bh = blockIdx.y;
    const int b  = bh / TA_NH;
    const int h  = bh % TA_NH;
    const int q0 = blockIdx.x * 128 + wave * 16;
    const float scale = 0.125f;  // HD^-0.5

    const int skey = tid >> 3, sd = (tid & 7) * 8;   // K chunk coords
    const int svd = tid >> 2, sko = (tid & 3) * 8;   // V chunk coords

    auto stage = [&](int buf, int kc) {
        async_b128(lds_addr(&Ks[buf][skey * TA_HD + sd]),
                   Kb + (size_t)(b * TA_S + kc + skey) * TA_H + h * TA_HD + sd);
        async_b128(lds_addr(&Vs[buf][svd * AKC + sko]),
                   Vt + (size_t)(h * TA_HD + svd) * TA_M + b * TA_S + kc + sko);
    };

    // Q fragments in A layout, two 32-wide d slices
    v16bf qa[2];
    {
        size_t rowQ = (size_t)(b * TA_S + q0 + laneM) * TA_H + h * TA_HD;
        #pragma unroll
        for (int f = 0; f < 2; ++f) {
            const unsigned short* p = Qb + rowQ + f * 32;
            qa[f] = load_frag2(p + laneHi * 8, p + 16 + laneHi * 8);
        }
    }

    v8f o[4];
    #pragma unroll
    for (int t = 0; t < 4; ++t) o[t] = (v8f){};
    float mrow[8], lrow[8];
    #pragma unroll
    for (int r = 0; r < 8; ++r) { mrow[r] = -1e30f; lrow[r] = 0.0f; }

    stage(0, 0);
    int buf = 0;
    for (int kc = 0; kc < TA_S; kc += AKC) {
        wait_async0();
        __syncthreads();
        if (kc + AKC < TA_S) stage(buf ^ 1, kc + AKC);

        // --- S = Q K^T: two 16-key tiles x two 32-d slices ------------------
        v8f sfrag[2];
        #pragma unroll
        for (int j = 0; j < 2; ++j) {
            v8f s = (v8f){};
            #pragma unroll
            for (int f = 0; f < 2; ++f) {
                const unsigned short* p =
                    &Ks[buf][(j * 16 + laneM) * TA_HD + f * 32];
                s = wmma_bf16(qa[f], load_frag2(p + laneHi * 8, p + 16 + laneHi * 8), s);
            }
            sfrag[j] = s;
        }
        #pragma unroll
        for (int j = 0; j < 2; ++j)
            #pragma unroll
            for (int r = 0; r < 8; ++r) sfrag[j][r] *= scale;

        // --- online softmax: DPP16 row butterflies --------------------------
        #pragma unroll
        for (int r = 0; r < 8; ++r) {
            float v = rowmax16(fmaxf(sfrag[0][r], sfrag[1][r]));
            float mnew  = fmaxf(mrow[r], v);
            float alpha = __expf(mrow[r] - mnew);
            float p0 = __expf(sfrag[0][r] - mnew);
            float p1 = __expf(sfrag[1][r] - mnew);
            sfrag[0][r] = p0; sfrag[1][r] = p1;
            float rs = rowsum16(p0 + p1);
            lrow[r] = lrow[r] * alpha + rs;
            mrow[r] = mnew;
            #pragma unroll
            for (int t = 0; t < 4; ++t) o[t][r] *= alpha;
        }

        // --- C-layout P -> per-wave LDS -> A-layout fragment ----------------
        #pragma unroll
        for (int j = 0; j < 2; ++j)
            #pragma unroll
            for (int r = 0; r < 8; ++r)
                Ps[wave][(laneHi * 8 + r) * AKC + j * 16 + laneM] =
                    f32_to_bf16(sfrag[j][r]);
        wait_ds0();
        const unsigned short* pp = &Ps[wave][laneM * AKC];
        v16bf pa = load_frag2(pp + laneHi * 8, pp + 16 + laneHi * 8);

        // --- O += P V -------------------------------------------------------
        #pragma unroll
        for (int t = 0; t < 4; ++t) {
            const unsigned short* vp = &Vs[buf][(t * 16 + laneM) * AKC];
            o[t] = wmma_bf16(pa, load_frag2(vp + laneHi * 8, vp + 16 + laneHi * 8), o[t]);
        }

        __syncthreads();
        buf ^= 1;
    }

    // --- normalize and write bf16 attention output --------------------------
    #pragma unroll
    for (int r = 0; r < 8; ++r) {
        float inv = 1.0f / lrow[r];
        size_t row = (size_t)(b * TA_S + q0 + laneHi * 8 + r) * TA_H + h * TA_HD;
        #pragma unroll
        for (int t = 0; t < 4; ++t)
            Ob[row + t * 16 + laneM] = f32_to_bf16(o[t][r] * inv);
    }
}

// ---------------------------------------------------------------------------
// Host launcher
// ---------------------------------------------------------------------------
extern "C" void kernel_launch(void* const* d_in, const int* in_sizes, int n_in,
                              void* d_out, int out_size, void* d_ws, size_t ws_size,
                              hipStream_t stream) {
    (void)in_sizes; (void)n_in; (void)out_size; (void)ws_size;
    const int M = TA_M, H = TA_H, R = TA_R;

    const float* hs  = (const float*)d_in[0];
    const float* Wq  = (const float*)d_in[1];
    const float* bq  = (const float*)d_in[2];
    const float* Wk  = (const float*)d_in[3];
    const float* bk  = (const float*)d_in[4];
    const float* Wv  = (const float*)d_in[5];
    const float* bv  = (const float*)d_in[6];
    const float* Wo  = (const float*)d_in[7];
    const float* bo  = (const float*)d_in[8];
    const float* qta = (const float*)d_in[9];
    const float* qtb = (const float*)d_in[10];
    const float* kta = (const float*)d_in[11];
    const float* ktb = (const float*)d_in[12];
    const float* cpl = (const float*)d_in[13];
    float* out = (float*)d_out;

    // ---- workspace carving -------------------------------------------------
    char* w = (char*)d_ws;
    size_t off = 0;
    auto carve = [&](size_t bytes) { char* p = w + off; off += (bytes + 255) & ~(size_t)255; return p; };
    unsigned short* Xb   = (unsigned short*)carve((size_t)M * H * 2);
    unsigned short* WqT  = (unsigned short*)carve((size_t)H * H * 2);
    unsigned short* WkT  = (unsigned short*)carve((size_t)H * H * 2);
    unsigned short* WvT  = (unsigned short*)carve((size_t)H * H * 2);
    unsigned short* WoT  = (unsigned short*)carve((size_t)H * H * 2);
    unsigned short* qaT  = (unsigned short*)carve((size_t)2 * R * H * 2);  // [2R][H]
    unsigned short* qbT  = (unsigned short*)carve((size_t)2 * H * R * 2);  // [H][R] x2
    unsigned short* kaT  = (unsigned short*)carve((size_t)2 * R * H * 2);
    unsigned short* kbT  = (unsigned short*)carve((size_t)2 * H * R * 2);
    float*          Qf   = (float*)carve((size_t)M * H * 4);
    float*          Kf   = (float*)carve((size_t)M * H * 4);
    unsigned short* Qbm  = (unsigned short*)carve((size_t)M * H * 2);
    unsigned short* Kbm  = (unsigned short*)carve((size_t)M * H * 2);
    unsigned short* VtT  = (unsigned short*)carve((size_t)H * M * 2);      // [H][M]
    unsigned short* Low  = (unsigned short*)carve((size_t)M * 2 * R * 2);  // [M][128]
    unsigned short* Obm  = (unsigned short*)carve((size_t)M * H * 2);

    // ---- 1. casts ----------------------------------------------------------
    ta_cast_f32_bf16<<<(M * H + 255) / 256, 256, 0, stream>>>(hs, Xb, M * H);
    struct { const float* s; unsigned short* d; int K, N; } tr[] = {
        { Wq, WqT, H, H }, { Wk, WkT, H, H }, { Wv, WvT, H, H }, { Wo, WoT, H, H },
        { qta,         qaT,         H, R }, { qta + H * R, qaT + R * H, H, R },
        { qtb,         qbT,         R, H }, { qtb + R * H, qbT + H * R, R, H },
        { kta,         kaT,         H, R }, { kta + H * R, kaT + R * H, H, R },
        { ktb,         kbT,         R, H }, { ktb + R * H, kbT + H * R, R, H },
    };
    for (auto& t : tr)
        ta_cast_transpose<<<(t.K * t.N + 255) / 256, 256, 0, stream>>>(t.s, t.d, t.K, t.N);

    const dim3 blk(256);
    const dim3 gFull(M / GBM, H / GBN);   // 32 x 16
    const dim3 gLow(M / GBM, 2 * R / GBN);// 32 x 2 (both torsion orders at once)

    // ---- 2. QKV projections ------------------------------------------------
    ta_gemm_bf16<true, true, true, false><<<gFull, blk, 0, stream>>>(
        Xb, WqT, bq, Qf, Qbm, nullptr, M, H, H);
    ta_gemm_bf16<true, true, true, false><<<gFull, blk, 0, stream>>>(
        Xb, WkT, bk, Kf, Kbm, nullptr, M, H, H);
    ta_gemm_bf16<true, false, false, true><<<gFull, blk, 0, stream>>>(
        Xb, WvT, bv, nullptr, nullptr, VtT, M, H, H);   // V written transposed

    // ---- 3. torsion on q: one N=128 low GEMM + fused lin/pointwise ---------
    ta_gemm_bf16<false, false, true, false><<<gLow, blk, 0, stream>>>(
        Qbm, qaT, nullptr, nullptr, Low, nullptr, M, 2 * R, H);
    ta_torsion_fused<<<gFull, blk, 0, stream>>>(
        Low, qbT, qbT + H * R, Qf, cpl, Qbm);

    // ---- 4. torsion on k ---------------------------------------------------
    ta_gemm_bf16<false, false, true, false><<<gLow, blk, 0, stream>>>(
        Kbm, kaT, nullptr, nullptr, Low, nullptr, M, 2 * R, H);
    ta_torsion_fused<<<gFull, blk, 0, stream>>>(
        Low, kbT, kbT + H * R, Kf, cpl, Kbm);

    // ---- 5. flash attention ------------------------------------------------
    ta_flash_attn<<<dim3(TA_S / 128, TA_B * TA_NH), blk, 0, stream>>>(
        Qbm, Kbm, VtT, Obm);

    // ---- 6. output projection ----------------------------------------------
    ta_gemm_bf16<true, true, false, false><<<gFull, blk, 0, stream>>>(
        Obm, WoT, bo, out, nullptr, nullptr, M, H, H);
}
